// SKOLRTransitionModel_30305289240701
// MI455X (gfx1250) — matrix operand order repaired and checked by reading, not compiled, not run on hardware
//
#include <hip/hip_runtime.h>
#include <hip/hip_bf16.h>
#include <math.h>

// ---- problem constants ----
constexpr int B_SZ = 16384, D = 256, U = 32, NBR = 8, MH = 512, RH = 256, NY = 20;
constexpr int RHU = RH + U;         // 288
constexpr int BT  = 64;             // batch rows per workgroup

typedef __attribute__((ext_vector_type(16))) __bf16 v16bf;
typedef __attribute__((ext_vector_type(8)))  float  v8f;

// ---- LDS layout (bytes) ----
constexpr int ZT_STR  = 264;  // bf16 elems per row (pad vs 256); row = 528 B (16B aligned)
constexpr int UT_STR  = 40;   // row = 80 B
constexpr int XB_STR  = 520;  // row = 1040 B
constexpr int GB_STR  = 264;
constexpr int ZO_STR  = 260;  // f32 epilogue tile (overlays XBF); row = 1040 B
constexpr int OFF_ZT   = 0;
constexpr int OFF_UT   = OFF_ZT  + BT * ZT_STR * 2;     //  33792
constexpr int OFF_XBF  = OFF_UT  + BT * UT_STR * 2;     //  38912
constexpr int OFF_GBF  = OFF_XBF + BT * XB_STR * 2;     // 105472
constexpr int OFF_BINP = OFF_GBF + BT * GB_STR * 2;     // 139264
constexpr int OFF_REDS = OFF_BINP + BT * GB_STR * 2;    // 173056
constexpr int OFF_REDQ = OFF_REDS + 8 * BT * 4;         // 175104
constexpr int OFF_MU   = OFF_REDQ + 8 * BT * 4;         // 177152
constexpr int OFF_RI   = OFF_MU + BT * 4;               // 177408
constexpr int SMEM_BYTES = OFF_RI + BT * 4;             // 177664

// ---- workspace (bf16 weights) offsets in d_ws, bytes ----
constexpr size_t WS_W1G = 0;                                  // [NBR][MH][D]
constexpr size_t WS_W2  = WS_W1G + (size_t)NBR * MH * D * 2;  // [NBR][RH][MH]
constexpr size_t WS_BB  = WS_W2  + (size_t)NBR * RH * MH * 2; // [NBR][RH][RHU]
constexpr size_t WS_WO  = WS_BB  + (size_t)NBR * RH * RHU * 2;// [NBR][D][RH]

// ================= fragment helpers (CDNA5 wave32 WMMA layouts) =============
// A (16x32 bf16): lane L: row m=L&15, half h=L>>4.
// VGPRs 0-3 hold k = {0..7}+8h  (16 contiguous bytes),
// VGPRs 4-7 hold k = {16..23}+8h (16 contiguous bytes) -> two ds_load_b128.
__device__ __forceinline__ v16bf frag_a_lds(const __bf16* base, int stride,
                                            int row0, int k0) {
  const int lane = threadIdx.x & 31;
  const int m = lane & 15, h = lane >> 4;
  const __bf16* rp = base + (row0 + m) * stride + k0 + 8 * h;
  union { v16bf v; uint4 q[2]; } f;
  f.q[0] = *(const uint4*)(rp);
  f.q[1] = *(const uint4*)(rp + 16);
  return f.v;
}
// B (32x16 bf16) from row-major weights W[N][K]: lane L: col n=L&15,
// same k order as A: k = {0..7}+8h and {16..23}+8h -> two 16-byte loads
__device__ __forceinline__ v16bf frag_b_glb(const __bf16* base, int stride,
                                            int n0, int k0) {
  const int lane = threadIdx.x & 31;
  const int n = lane & 15, kb = (lane >> 4) * 8;
  const __bf16* p = base + (size_t)(n0 + n) * stride + k0 + kb;
  union { v16bf v; uint4 q[2]; } f;
  f.q[0] = *(const uint4*)(p);
  f.q[1] = *(const uint4*)(p + 16);
  return f.v;
}

__device__ __forceinline__ v8f wmma_bf16(v16bf a, v16bf b, v8f c) {
  return __builtin_amdgcn_wmma_f32_16x16x32_bf16(false, a, false, b,
                                                 (short)0, c, false, false);
}

__device__ __forceinline__ float gelu_exact(float x) {
  return 0.5f * x * (1.0f + erff(x * 0.70710678118654752f));
}

// ========================= convert kernels ==================================
__global__ void cvt_w1_gate(const float* __restrict__ W1,
                            const float* __restrict__ gates,
                            __bf16* __restrict__ out) {
  int i = blockIdx.x * blockDim.x + threadIdx.x;
  if (i >= NBR * MH * D) return;
  int d = i & (D - 1);
  int n = i / (MH * D);
  float g = gates[n * D + d];
  float s = 1.0f / (1.0f + __expf(-g));
  out[i] = (__bf16)(W1[i] * s);
}

__global__ void cvt_f32_bf16(const float* __restrict__ in,
                             __bf16* __restrict__ out, int nelem) {
  int i = blockIdx.x * blockDim.x + threadIdx.x;
  if (i < nelem) out[i] = (__bf16)in[i];
}

// ========================= fused main kernel ================================
__global__ void __launch_bounds__(256)
skolr_fused(const float* __restrict__ zt, const float* __restrict__ dtp,
            const float* __restrict__ ut, const float* __restrict__ b1,
            const float* __restrict__ lng, const float* __restrict__ lnb,
            const float* __restrict__ b2, const __bf16* __restrict__ W1g,
            const __bf16* __restrict__ W2b, const __bf16* __restrict__ Bb,
            const __bf16* __restrict__ Wob, const float* __restrict__ Cm,
            const float* __restrict__ Dmm, float* __restrict__ out) {
  extern __shared__ char smem[];
  __bf16* sZT  = (__bf16*)(smem + OFF_ZT);
  __bf16* sUT  = (__bf16*)(smem + OFF_UT);
  __bf16* sXB  = (__bf16*)(smem + OFF_XBF);
  __bf16* sGB  = (__bf16*)(smem + OFF_GBF);
  __bf16* sBI  = (__bf16*)(smem + OFF_BINP);
  float*  redS = (float*)(smem + OFF_REDS);
  float*  redQ = (float*)(smem + OFF_REDQ);
  float*  sMu  = (float*)(smem + OFF_MU);
  float*  sRi  = (float*)(smem + OFF_RI);

  const int tid  = threadIdx.x;
  const int w    = tid >> 5;
  const int lane = tid & 31;
  const int lm   = lane & 15;   // column within 16-tile (B/C/D layouts)
  const int lh   = lane >> 4;   // half select
  const int rowBase = blockIdx.x * BT;
  const float dtv = dtp[0];

  // ---- stage zt (bf16, packed b64 stores) and ut*dt (bf16) into LDS ----
  for (int it = tid; it < BT * D / 4; it += 256) {
    int r = it >> 6, c4 = it & 63;
    float4 vz = ((const float4*)(zt + (size_t)(rowBase + r) * D))[c4];
    union { __bf16 h[4]; uint2 u; } pk;
    pk.h[0] = (__bf16)vz.x; pk.h[1] = (__bf16)vz.y;
    pk.h[2] = (__bf16)vz.z; pk.h[3] = (__bf16)vz.w;
    *(uint2*)(sZT + r * ZT_STR + c4 * 4) = pk.u;
  }
  for (int it = tid; it < BT * U; it += 256) {
    int r = it >> 5, c = it & 31;
    sUT[r * UT_STR + c] = (__bf16)(ut[(size_t)(rowBase + r) * U + c] * dtv);
  }
  __syncthreads();

  const v8f vzero = {0.f, 0.f, 0.f, 0.f, 0.f, 0.f, 0.f, 0.f};
  v8f accz[4][2];
#pragma unroll
  for (int mt = 0; mt < 4; ++mt)
#pragma unroll
    for (int nt = 0; nt < 2; ++nt) accz[mt][nt] = vzero;

  for (int n = 0; n < NBR; ++n) {
    // ---------------- GEMM1: x[64x512] = ztile @ W1g_n^T ----------------
    const __bf16* W1n = W1g + (size_t)n * MH * D;
    const int hBase = w * 64;
    v8f acc1[4][4];
#pragma unroll
    for (int mt = 0; mt < 4; ++mt)
#pragma unroll
      for (int nt = 0; nt < 4; ++nt) acc1[mt][nt] = vzero;
    for (int kb = 0; kb < D; kb += 32) {
      __builtin_prefetch(W1n + (size_t)hBase * D + kb + 128, 0, 1);
      v16bf a[4], b[4];
#pragma unroll
      for (int mt = 0; mt < 4; ++mt) a[mt] = frag_a_lds(sZT, ZT_STR, mt * 16, kb);
#pragma unroll
      for (int nt = 0; nt < 4; ++nt) b[nt] = frag_b_glb(W1n, D, hBase + nt * 16, kb);
#pragma unroll
      for (int mt = 0; mt < 4; ++mt)
#pragma unroll
        for (int nt = 0; nt < 4; ++nt)
          acc1[mt][nt] = wmma_bf16(a[mt], b[nt], acc1[mt][nt]);
    }
    // + b1 (column-dependent)
#pragma unroll
    for (int nt = 0; nt < 4; ++nt) {
      float bb = b1[(size_t)n * MH + hBase + nt * 16 + lm];
#pragma unroll
      for (int mt = 0; mt < 4; ++mt)
#pragma unroll
        for (int v = 0; v < 8; ++v) acc1[mt][nt][v] += bb;
    }
    // ---- LayerNorm stats straight from accumulators ----
#pragma unroll
    for (int mt = 0; mt < 4; ++mt)
#pragma unroll
      for (int v = 0; v < 8; ++v) {
        float s = acc1[mt][0][v] + acc1[mt][1][v] + acc1[mt][2][v] + acc1[mt][3][v];
        float q = acc1[mt][0][v] * acc1[mt][0][v] + acc1[mt][1][v] * acc1[mt][1][v] +
                  acc1[mt][2][v] * acc1[mt][2][v] + acc1[mt][3][v] * acc1[mt][3][v];
#pragma unroll
        for (int msk = 1; msk < 16; msk <<= 1) {
          s += __shfl_xor(s, msk, 32);
          q += __shfl_xor(q, msk, 32);
        }
        if (lm == 0) {
          int row = mt * 16 + v + 8 * lh;
          redS[w * BT + row] = s;
          redQ[w * BT + row] = q;
        }
      }
    __syncthreads();
    if (tid < BT) {
      float s = 0.f, q = 0.f;
      for (int ww = 0; ww < 8; ++ww) { s += redS[ww * BT + tid]; q += redQ[ww * BT + tid]; }
      float mu  = s * (1.0f / MH);
      float var = q * (1.0f / MH) - mu * mu;
      sMu[tid] = mu;
      sRi[tid] = rsqrtf(var + 1e-5f);
    }
    __syncthreads();
    // ---- apply LN + exact GELU, emit bf16 x-tile to LDS ----
#pragma unroll
    for (int nt = 0; nt < 4; ++nt) {
      int hcol = hBase + nt * 16 + lm;
      float g = lng[(size_t)n * MH + hcol];
      float bta = lnb[(size_t)n * MH + hcol];
#pragma unroll
      for (int mt = 0; mt < 4; ++mt)
#pragma unroll
        for (int v = 0; v < 8; ++v) {
          int row = mt * 16 + v + 8 * lh;
          float xv = (acc1[mt][nt][v] - sMu[row]) * sRi[row] * g + bta;
          sXB[row * XB_STR + hcol] = (__bf16)gelu_exact(xv);
        }
    }
    __syncthreads();
    // ---------------- GEMM2: g[64x256] = x @ W2_n^T ----------------
    const __bf16* W2n = W2b + (size_t)n * RH * MH;
    const int rBase = w * 32;
    v8f acc2[4][2];
#pragma unroll
    for (int mt = 0; mt < 4; ++mt)
#pragma unroll
      for (int nt = 0; nt < 2; ++nt) acc2[mt][nt] = vzero;
    for (int kb = 0; kb < MH; kb += 32) {
      __builtin_prefetch(W2n + (size_t)rBase * MH + kb + 128, 0, 1);
      v16bf a[4], b[2];
#pragma unroll
      for (int mt = 0; mt < 4; ++mt) a[mt] = frag_a_lds(sXB, XB_STR, mt * 16, kb);
#pragma unroll
      for (int nt = 0; nt < 2; ++nt) b[nt] = frag_b_glb(W2n, MH, rBase + nt * 16, kb);
#pragma unroll
      for (int mt = 0; mt < 4; ++mt)
#pragma unroll
        for (int nt = 0; nt < 2; ++nt)
          acc2[mt][nt] = wmma_bf16(a[mt], b[nt], acc2[mt][nt]);
    }
#pragma unroll
    for (int nt = 0; nt < 2; ++nt) {
      float bb = b2[(size_t)n * RH + rBase + nt * 16 + lm];
#pragma unroll
      for (int mt = 0; mt < 4; ++mt)
#pragma unroll
        for (int v = 0; v < 8; ++v) {
          int row = mt * 16 + v + 8 * lh;
          sGB[row * GB_STR + rBase + nt * 16 + lm] = (__bf16)(acc2[mt][nt][v] + bb);
        }
    }
    __syncthreads();
    // ---------- GEMM3: b_inp = g @ Bg_n^T + ut_dt @ Bu_n^T ----------
    const __bf16* Bn = Bb + (size_t)n * RH * RHU;
    v8f acc3[4][2];
#pragma unroll
    for (int mt = 0; mt < 4; ++mt)
#pragma unroll
      for (int nt = 0; nt < 2; ++nt) acc3[mt][nt] = vzero;
    for (int kb = 0; kb < RH; kb += 32) {
      v16bf a[4], b[2];
#pragma unroll
      for (int mt = 0; mt < 4; ++mt) a[mt] = frag_a_lds(sGB, GB_STR, mt * 16, kb);
#pragma unroll
      for (int nt = 0; nt < 2; ++nt) b[nt] = frag_b_glb(Bn, RHU, rBase + nt * 16, kb);
#pragma unroll
      for (int mt = 0; mt < 4; ++mt)
#pragma unroll
        for (int nt = 0; nt < 2; ++nt)
          acc3[mt][nt] = wmma_bf16(a[mt], b[nt], acc3[mt][nt]);
    }
    {  // control-input tail, K = U = 32
      v16bf a[4], b[2];
#pragma unroll
      for (int mt = 0; mt < 4; ++mt) a[mt] = frag_a_lds(sUT, UT_STR, mt * 16, 0);
#pragma unroll
      for (int nt = 0; nt < 2; ++nt) b[nt] = frag_b_glb(Bn, RHU, rBase + nt * 16, RH);
#pragma unroll
      for (int mt = 0; mt < 4; ++mt)
#pragma unroll
        for (int nt = 0; nt < 2; ++nt)
          acc3[mt][nt] = wmma_bf16(a[mt], b[nt], acc3[mt][nt]);
    }
#pragma unroll
    for (int nt = 0; nt < 2; ++nt)
#pragma unroll
      for (int mt = 0; mt < 4; ++mt)
#pragma unroll
        for (int v = 0; v < 8; ++v) {
          int row = mt * 16 + v + 8 * lh;
          sBI[row * GB_STR + rBase + nt * 16 + lm] = (__bf16)acc3[mt][nt][v];
        }
    __syncthreads();
    // ---------- GEMM4: zt1 += b_inp @ W_out_n^T (persistent acc) ----------
    const __bf16* Won = Wob + (size_t)n * D * RH;
    const int dBase = w * 32;
    for (int kb = 0; kb < RH; kb += 32) {
      v16bf a[4], b[2];
#pragma unroll
      for (int mt = 0; mt < 4; ++mt) a[mt] = frag_a_lds(sBI, GB_STR, mt * 16, kb);
#pragma unroll
      for (int nt = 0; nt < 2; ++nt) b[nt] = frag_b_glb(Won, RH, dBase + nt * 16, kb);
#pragma unroll
      for (int mt = 0; mt < 4; ++mt)
#pragma unroll
        for (int nt = 0; nt < 2; ++nt)
          accz[mt][nt] = wmma_bf16(a[mt], b[nt], accz[mt][nt]);
    }
    __syncthreads();  // LDS tiles reused next branch iteration
  }

  // ---- zt1 -> f32 LDS tile (overlays sXB), then coalesced global stores ----
  float* zout = (float*)(smem + OFF_XBF);  // [BT][ZO_STR]
  const int dBase = w * 32;
#pragma unroll
  for (int mt = 0; mt < 4; ++mt)
#pragma unroll
    for (int nt = 0; nt < 2; ++nt)
#pragma unroll
      for (int v = 0; v < 8; ++v) {
        int row = mt * 16 + v + 8 * lh;
        zout[row * ZO_STR + dBase + nt * 16 + lm] = accz[mt][nt][v];
      }
  __syncthreads();
  for (int it = tid; it < BT * D / 4; it += 256) {
    int r = it >> 6, c4 = it & 63;
    float4 vv = *(const float4*)(zout + r * ZO_STR + c4 * 4);
    ((float4*)(out + (size_t)(rowBase + r) * D))[c4] = vv;
  }
  // ---- yt1 = zt1 @ C^T + ut_dt @ Dm^T  (tiny: VALU) ----
  float* yout = out + (size_t)B_SZ * D;
  for (int it = tid; it < BT * NY; it += 256) {
    int r = it / NY, y = it % NY;
    float acc = 0.f;
    for (int d0 = 0; d0 < D; ++d0) acc += zout[r * ZO_STR + d0] * Cm[(size_t)y * D + d0];
    float au = 0.f;
    for (int u0 = 0; u0 < U; ++u0)
      au += ut[(size_t)(rowBase + r) * U + u0] * Dmm[(size_t)y * U + u0];
    yout[(size_t)(rowBase + r) * NY + y] = acc + au * dtv;
  }
}

// ============================== launch ======================================
extern "C" void kernel_launch(void* const* d_in, const int* in_sizes, int n_in,
                              void* d_out, int out_size, void* d_ws, size_t ws_size,
                              hipStream_t stream) {
  const float* zt    = (const float*)d_in[0];
  const float* dtp   = (const float*)d_in[1];
  const float* ut    = (const float*)d_in[2];
  const float* gates = (const float*)d_in[3];
  const float* W1    = (const float*)d_in[4];
  const float* b1    = (const float*)d_in[5];
  const float* lng   = (const float*)d_in[6];
  const float* lnb   = (const float*)d_in[7];
  const float* W2    = (const float*)d_in[8];
  const float* b2    = (const float*)d_in[9];
  // d_in[10], d_in[11]: lam_r, lam_i — drop out of the math (h0 = 0)
  const float* rnnB  = (const float*)d_in[12];
  const float* Wout  = (const float*)d_in[13];
  const float* Cm    = (const float*)d_in[14];
  const float* Dmm   = (const float*)d_in[15];
  float* out = (float*)d_out;

  __bf16* W1g = (__bf16*)((char*)d_ws + WS_W1G);
  __bf16* W2b = (__bf16*)((char*)d_ws + WS_W2);
  __bf16* Bbw = (__bf16*)((char*)d_ws + WS_BB);
  __bf16* Wob = (__bf16*)((char*)d_ws + WS_WO);

  cvt_w1_gate<<<(NBR * MH * D + 255) / 256, 256, 0, stream>>>(W1, gates, W1g);
  cvt_f32_bf16<<<(NBR * RH * MH + 255) / 256, 256, 0, stream>>>(W2, W2b, NBR * RH * MH);
  cvt_f32_bf16<<<(NBR * RH * RHU + 255) / 256, 256, 0, stream>>>(rnnB, Bbw, NBR * RH * RHU);
  cvt_f32_bf16<<<(NBR * D * RH + 255) / 256, 256, 0, stream>>>(Wout, Wob, NBR * D * RH);

  skolr_fused<<<B_SZ / BT, 256, SMEM_BYTES, stream>>>(
      zt, dtp, ut, b1, lng, lnb, b2, W1g, W2b, Bbw, Wob, Cm, Dmm, out);
}
